// GNNGenerator_18416819765911
// MI455X (gfx1250) — compile-verified, matrix operand order checked by reading, not run on hardware
//
#include <hip/hip_runtime.h>
#include <hip/hip_bf16.h>

#define NN 100000
#define EE 1600000
#define HH 4
#define DD 32

typedef __attribute__((ext_vector_type(16))) __bf16 v16bf;
typedef __attribute__((ext_vector_type(8)))  float  v8f;

union Frag {
    v16bf v;
    uint4 q[2];
};

__device__ __forceinline__ unsigned short f2bf(float f) {
    unsigned u = __float_as_uint(f);
    unsigned r = u + 0x7FFFu + ((u >> 16) & 1u);
    return (unsigned short)(r >> 16);
}

__device__ __forceinline__ unsigned orderf(float f) {
    unsigned u = __float_as_uint(f);
    return (u & 0x80000000u) ? ~u : (u | 0x80000000u);
}

__device__ __forceinline__ float unorderf(unsigned v) {
    return (v & 0x80000000u) ? __uint_as_float(v ^ 0x80000000u)
                             : __uint_as_float(~v);
}

// ---------------------------------------------------------------------------
// 1a) Transpose 4 weight matrices (128x128, row-major KxN) -> bf16 (NxK).
//     WqT/WkT/WvT land contiguously (single base pointer for k_qkv).
// ---------------------------------------------------------------------------
__global__ void k_prep(const float* __restrict__ Wq, const float* __restrict__ Wk,
                       const float* __restrict__ Wv, const float* __restrict__ Wo,
                       unsigned short* __restrict__ WT3,    // [3][128][128] bf16
                       unsigned short* __restrict__ WoT) {  // [128][128]   bf16
    int tid = blockIdx.x * 256 + threadIdx.x;          // 0 .. 65535
    int mat = tid >> 14;
    int idx = tid & 16383;
    int k = idx >> 7;
    int n = idx & 127;
    const float* src = (mat == 0) ? Wq : (mat == 1) ? Wk : (mat == 2) ? Wv : Wo;
    unsigned short val = f2bf(src[k * 128 + n]);
    if (mat < 3)
        WT3[mat * 16384 + n * 128 + k] = val;
    else
        WoT[n * 128 + k] = val;
}

// 1b) Pack biases contiguously: biasAll = [bq | bk | bv | bo] (4*128 f32)
__global__ void k_prep_bias(const float* __restrict__ bq, const float* __restrict__ bk,
                            const float* __restrict__ bv, const float* __restrict__ bo,
                            float* __restrict__ biasAll) {
    int tid = blockIdx.x * 256 + threadIdx.x;          // 0 .. 511
    int mat = tid >> 7;
    int n = tid & 127;
    const float* src = (mat == 0) ? bq : (mat == 1) ? bk : (mat == 2) ? bv : bo;
    biasAll[tid] = src[n];
}

// ---------------------------------------------------------------------------
// 2) Zero a buffer (segmax bits / denom / agg)
// ---------------------------------------------------------------------------
__global__ void k_zero(unsigned* __restrict__ p, long n) {
    long i = (long)blockIdx.x * 256 + threadIdx.x;
    if (i < n) p[i] = 0u;
}

// ---------------------------------------------------------------------------
// 3) QKV projection: per block 16 nodes, 8 waves x 3 col-tiles, K=128 in 4 steps.
//    Single base pointers (WT3, QKV, biasAll) keep loads/stores in the global
//    address space -> global_load_b128 (no flat/DScnt entanglement).
// ---------------------------------------------------------------------------
__global__ void __launch_bounds__(256, 2)
k_qkv(const float* __restrict__ nf, const float* __restrict__ qe,
      const unsigned short* __restrict__ WT3,   // [3][128][128] bf16, N-major
      const float* __restrict__ biasAll,        // [4][128] (bq,bk,bv,bo)
      float* __restrict__ QKV) {                // [3][NN][128] f32
    __shared__ __align__(16) unsigned short sE[16 * 136];   // enh tile, bf16, padded

    const int t = threadIdx.x;
    const int lane = t & 31;
    const int wave = t >> 5;
    const int node0 = blockIdx.x * 16;

    // speculative prefetch of next tile's node rows
    if (node0 + 16 < NN) {
        const float* nxt = nf + (size_t)(node0 + 16) * 128 + (t & 15) * 128 / 16;
        __builtin_prefetch(nxt, 0, 1);
    }

    // stage enh = nf + 0.1*query into LDS as bf16
#pragma unroll
    for (int i = 0; i < 2; ++i) {
        int idx4 = t * 2 + i;            // 16 rows x 32 float4
        int rr = idx4 >> 5;
        int c4 = idx4 & 31;
        float4 v = ((const float4*)(nf + (size_t)(node0 + rr) * 128))[c4];
        float4 q = ((const float4*)qe)[c4];
        int base = rr * 136 + c4 * 4;
        sE[base + 0] = f2bf(v.x + 0.1f * q.x);
        sE[base + 1] = f2bf(v.y + 0.1f * q.y);
        sE[base + 2] = f2bf(v.z + 0.1f * q.z);
        sE[base + 3] = f2bf(v.w + 0.1f * q.w);
    }
    __syncthreads();

    v8f acc[3];
#pragma unroll
    for (int j = 0; j < 3; ++j) acc[j] = (v8f){0, 0, 0, 0, 0, 0, 0, 0};

#pragma unroll
    for (int kk = 0; kk < 4; ++kk) {
        const int kb = 32 * kk + ((lane & 16) ? 8 : 0);
        Frag a;
        const unsigned short* arow = sE + (lane & 15) * 136 + kb;
        a.q[0] = *(const uint4*)(arow);
        a.q[1] = *(const uint4*)(arow + 16);
#pragma unroll
        for (int j = 0; j < 3; ++j) {
            const int ct = wave + 8 * j;         // 0..23
            const int mat = ct >> 3;
            const int n0 = (ct & 7) * 16;
            const unsigned short* brow =
                WT3 + (size_t)mat * 16384 + (size_t)(n0 + (lane & 15)) * 128 + kb;
            Frag b;
            b.q[0] = *(const uint4*)(brow);
            b.q[1] = *(const uint4*)(brow + 16);
            acc[j] = __builtin_amdgcn_wmma_f32_16x16x32_bf16(
                false, a.v, false, b.v, (short)0, acc[j], false, false);
        }
    }

    const int mbase = (lane & 16) ? 8 : 0;
#pragma unroll
    for (int j = 0; j < 3; ++j) {
        const int ct = wave + 8 * j;
        const int mat = ct >> 3;
        const int n0 = (ct & 7) * 16;
        const int n = n0 + (lane & 15);
        const float bb = biasAll[mat * 128 + n];
        float* outp = QKV + (size_t)mat * NN * 128;
#pragma unroll
        for (int r = 0; r < 8; ++r) {
            int row = node0 + mbase + r;
            outp[(size_t)row * 128 + n] = acc[j][r] + bb;
        }
    }
}

// ---------------------------------------------------------------------------
// 4) Edge scores + atomic segment max (orderable uint encoding)
// ---------------------------------------------------------------------------
__global__ void k_edge_scores(const int* __restrict__ ei, const int* __restrict__ er,
                              const float* __restrict__ Q, const float* __restrict__ K,
                              const float* __restrict__ rel,
                              float* __restrict__ scores, unsigned* __restrict__ segmax) {
    int e = blockIdx.x * 256 + threadIdx.x;
    if (e >= EE) return;
    const int s = ei[e];
    const int d = ei[EE + e];
    const int r = er[e];
    const float4* qp = (const float4*)(Q + (size_t)d * 128);
    const float4* kp = (const float4*)(K + (size_t)s * 128);
    const float4* rp = (const float4*)(rel + (size_t)r * 32);
    const float invScale = 0.17677669529663687f;   // 1/sqrt(32)

#pragma unroll
    for (int h = 0; h < HH; ++h) {
        float acc = 0.f;
#pragma unroll
        for (int i = 0; i < 8; ++i) {
            float4 q = qp[h * 8 + i];
            float4 k = kp[h * 8 + i];
            float4 rr = rp[i];
            acc += q.x * (k.x + 0.2f * rr.x);
            acc += q.y * (k.y + 0.2f * rr.y);
            acc += q.z * (k.z + 0.2f * rr.z);
            acc += q.w * (k.w + 0.2f * rr.w);
        }
        float sc = acc * invScale;
        scores[(size_t)e * HH + h] = sc;
        atomicMax(segmax + (size_t)d * HH + h, orderf(sc));
    }
}

// ---------------------------------------------------------------------------
// 5) exp + atomic denom + unnormalized V aggregation
// ---------------------------------------------------------------------------
__global__ void k_edge_accum(const int* __restrict__ ei,
                             const float* __restrict__ scores,
                             const unsigned* __restrict__ segmax,
                             const float* __restrict__ V,
                             float* __restrict__ denom, float* __restrict__ agg) {
    long id = (long)blockIdx.x * 256 + threadIdx.x;
    long e = id >> 2;
    int h = (int)(id & 3);
    if (e >= EE) return;
    const int s = ei[e];
    const int d = ei[EE + e];
    const float sc = scores[e * HH + h];
    const float m = unorderf(segmax[(size_t)d * HH + h]);
    const float ex = __expf(sc - m);
    atomicAdd(denom + (size_t)d * HH + h, ex);
    const float* vp = V + (size_t)s * 128 + h * DD;
    float* ag = agg + (size_t)d * 128 + h * DD;
#pragma unroll
    for (int i = 0; i < DD; ++i) atomicAdd(ag + i, ex * vp[i]);
}

// ---------------------------------------------------------------------------
// 6) Output: msg -> bf16 LDS, WMMA with WoT, + bo + residual, LayerNorm
// ---------------------------------------------------------------------------
__global__ void __launch_bounds__(256, 2)
k_output(const float* __restrict__ nf,
         const float* __restrict__ denom, const float* __restrict__ agg,
         const float* __restrict__ V, const unsigned short* __restrict__ WoT,
         const float* __restrict__ biasAll,      // bo at offset 3*128
         const float* __restrict__ lng,
         const float* __restrict__ lnb, float* __restrict__ out) {
    __shared__ __align__(16) unsigned short sM[16 * 136];   // msg tile bf16
    __shared__ float sX[16 * 132];                          // pre-LN rows f32

    const int t = threadIdx.x;
    const int lane = t & 31;
    const int wave = t >> 5;
    const int node0 = blockIdx.x * 16;

    if (node0 + 16 < NN) {
        __builtin_prefetch(agg + (size_t)(node0 + 16) * 128 + t * 8, 0, 1);
    }

    // stage msg = has_in ? agg/denom : V  (bf16)
#pragma unroll
    for (int i = 0; i < 2; ++i) {
        int idx4 = t * 2 + i;
        int rr = idx4 >> 5;
        int c4 = idx4 & 31;
        int node = node0 + rr;
        int h = c4 >> 3;
        float den = denom[(size_t)node * HH + h];
        float4 a = ((const float4*)(agg + (size_t)node * 128))[c4];
        float4 vv = ((const float4*)(V + (size_t)node * 128))[c4];
        float mx, my, mz, mw;
        if (den > 0.f) {
            float inv = 1.0f / den;
            mx = a.x * inv; my = a.y * inv; mz = a.z * inv; mw = a.w * inv;
        } else {
            mx = vv.x; my = vv.y; mz = vv.z; mw = vv.w;
        }
        int base = rr * 136 + c4 * 4;
        sM[base + 0] = f2bf(mx);
        sM[base + 1] = f2bf(my);
        sM[base + 2] = f2bf(mz);
        sM[base + 3] = f2bf(mw);
    }
    __syncthreads();

    v8f acc = (v8f){0, 0, 0, 0, 0, 0, 0, 0};
#pragma unroll
    for (int kk = 0; kk < 4; ++kk) {
        const int kb = 32 * kk + ((lane & 16) ? 8 : 0);
        Frag a;
        const unsigned short* arow = sM + (lane & 15) * 136 + kb;
        a.q[0] = *(const uint4*)(arow);
        a.q[1] = *(const uint4*)(arow + 16);
        const unsigned short* brow = WoT + (size_t)(wave * 16 + (lane & 15)) * 128 + kb;
        Frag b;
        b.q[0] = *(const uint4*)(brow);
        b.q[1] = *(const uint4*)(brow + 16);
        acc = __builtin_amdgcn_wmma_f32_16x16x32_bf16(
            false, a.v, false, b.v, (short)0, acc, false, false);
    }

    // scatter C tile into f32 LDS with bias + residual
    const int n = wave * 16 + (lane & 15);
    const float add = biasAll[3 * 128 + n];
    const int mbase = (lane & 16) ? 8 : 0;
#pragma unroll
    for (int r = 0; r < 8; ++r) {
        int row = mbase + r;
        int node = node0 + row;
        sX[row * 132 + n] = acc[r] + add + nf[(size_t)node * 128 + n];
    }
    __syncthreads();

    // LayerNorm: 16 threads per row, half-wave butterfly reduction
    const int rr = t >> 4;
    const int p = t & 15;
    float s1 = 0.f, s2 = 0.f;
#pragma unroll
    for (int i = 0; i < 8; ++i) {
        float x = sX[rr * 132 + p * 8 + i];
        s1 += x;
        s2 += x * x;
    }
#pragma unroll
    for (int off = 8; off >= 1; off >>= 1) {
        s1 += __shfl_xor(s1, off, 32);
        s2 += __shfl_xor(s2, off, 32);
    }
    const float mu = s1 * (1.0f / 128.0f);
    const float var = s2 * (1.0f / 128.0f) - mu * mu;
    const float rs = rsqrtf(var + 1e-5f);
    const int node = node0 + rr;
#pragma unroll
    for (int i = 0; i < 8; ++i) {
        int c = p * 8 + i;
        float x = sX[rr * 132 + c];
        out[(size_t)node * 128 + c] = (x - mu) * rs * lng[c] + lnb[c];
    }
}

// ---------------------------------------------------------------------------
// Launch
// ---------------------------------------------------------------------------
extern "C" void kernel_launch(void* const* d_in, const int* in_sizes, int n_in,
                              void* d_out, int out_size, void* d_ws, size_t ws_size,
                              hipStream_t stream) {
    const float* nf  = (const float*)d_in[0];
    const int*   ei  = (const int*)d_in[1];
    const int*   er  = (const int*)d_in[2];
    const float* qe  = (const float*)d_in[3];
    const float* Wq  = (const float*)d_in[4];
    const float* bq  = (const float*)d_in[5];
    const float* Wk  = (const float*)d_in[6];
    const float* bk  = (const float*)d_in[7];
    const float* Wv  = (const float*)d_in[8];
    const float* bv  = (const float*)d_in[9];
    const float* rel = (const float*)d_in[10];
    const float* Wo  = (const float*)d_in[11];
    const float* bo  = (const float*)d_in[12];
    const float* lng = (const float*)d_in[13];
    const float* lnb = (const float*)d_in[14];
    float* out = (float*)d_out;

    char* ws = (char*)d_ws;
    const size_t OFF_WT3 = 0;                                // 3*128*128*2 = 98304
    const size_t OFF_WOT = OFF_WT3 + 98304;                  // 32768
    const size_t OFF_B   = OFF_WOT + 32768;                  // 4*128*4 = 2048
    const size_t OFF_Q   = OFF_B + 2048;                     // QKV contiguous
    const size_t OFF_AGG = OFF_Q + (size_t)3 * NN * 128 * 4;
    const size_t OFF_SC  = OFF_AGG + (size_t)NN * 128 * 4;
    const size_t OFF_MAX = OFF_SC + (size_t)EE * HH * 4;
    const size_t OFF_DEN = OFF_MAX + (size_t)NN * HH * 4;

    unsigned short* WT3 = (unsigned short*)(ws + OFF_WT3);
    unsigned short* WoT = (unsigned short*)(ws + OFF_WOT);
    float* biasAll = (float*)(ws + OFF_B);
    float* QKV    = (float*)(ws + OFF_Q);
    float* Q      = QKV;
    float* K      = QKV + (size_t)NN * 128;
    float* V      = QKV + (size_t)2 * NN * 128;
    float* agg    = (float*)(ws + OFF_AGG);
    float* scores = (float*)(ws + OFF_SC);
    unsigned* segmax = (unsigned*)(ws + OFF_MAX);
    float* denom  = (float*)(ws + OFF_DEN);

    // 1) weight transpose/convert + bias packing
    k_prep<<<256, 256, 0, stream>>>(Wq, Wk, Wv, Wo, WT3, WoT);
    k_prep_bias<<<2, 256, 0, stream>>>(bq, bk, bv, bo, biasAll);

    // 2) zero accumulators
    {
        long n_max = (long)NN * HH;
        long n_den = (long)NN * HH;
        long n_agg = (long)NN * 128;
        k_zero<<<(unsigned)((n_max + 255) / 256), 256, 0, stream>>>(segmax, n_max);
        k_zero<<<(unsigned)((n_den + 255) / 256), 256, 0, stream>>>((unsigned*)denom, n_den);
        k_zero<<<(unsigned)((n_agg + 255) / 256), 256, 0, stream>>>((unsigned*)agg, n_agg);
    }

    // 3) QKV projection (N/16 = 6250 tiles)
    k_qkv<<<NN / 16, 256, 0, stream>>>(nf, qe, WT3, biasAll, QKV);

    // 4) edge scores + segment max
    k_edge_scores<<<(EE + 255) / 256, 256, 0, stream>>>(ei, er, Q, K, rel, scores, segmax);

    // 5) exp + denom + aggregation
    k_edge_accum<<<(unsigned)(((long)EE * HH + 255) / 256), 256, 0, stream>>>(
        ei, scores, segmax, V, denom, agg);

    // 6) output projection + residual + LayerNorm
    k_output<<<NN / 16, 256, 0, stream>>>(nf, denom, agg, V, WoT, biasAll, lng, lnb, out);
}
